// ContactMapPredictor_32229434589572
// MI455X (gfx1250) — matrix operand order repaired and checked by reading, not compiled, not run on hardware
//
#include <hip/hip_runtime.h>

typedef _Float16 v16h __attribute__((ext_vector_type(16)));
typedef _Float16 v8h  __attribute__((ext_vector_type(8)));
typedef float    v8f  __attribute__((ext_vector_type(8)));
typedef float    v4f  __attribute__((ext_vector_type(4)));
typedef unsigned int v4u __attribute__((ext_vector_type(4)));
typedef int      v8i  __attribute__((ext_vector_type(8)));
typedef int      v4i  __attribute__((ext_vector_type(4)));

#define WMMA_F16(a, b, c) \
    __builtin_amdgcn_wmma_f32_16x16x32_f16(false, (a), false, (b), (short)0, (c), false, false)

// xor-swizzle a float across lanes (ds_swizzle group-of-32: and=0x1f, or=0, xor=m)
#define SWZ_XOR_F(v, m) \
    __int_as_float(__builtin_amdgcn_ds_swizzle(__float_as_int(v), (((m) << 10) | 0x1f)))

#define BATCH 16
#define SEQ   2048
#define EMB   768
#define HEADS 8
#define DH    96
#define MROWS (BATCH * SEQ)   // 32768

#if __has_builtin(__builtin_amdgcn_tensor_load_to_lds)
#define HAVE_TDM 1
#else
#define HAVE_TDM 0
#endif

// ---------------------------------------------------------------- cast f32->f16
__global__ void cast_f32_to_f16(const float* __restrict__ in,
                                _Float16* __restrict__ out, int n) {
    int i = blockIdx.x * blockDim.x + threadIdx.x;
    int stride = gridDim.x * blockDim.x;
    for (; i < n; i += stride) out[i] = (_Float16)in[i];
}

#if HAVE_TDM
// Issue one TDM descriptor: 2-D tile, tile_dim0=32 (K, fastest), tile_dim1=64 (cols),
// element = 2 bytes, row stride = K elements.  Fill order packs each column's 32
// k-values contiguously in LDS: Blds[col][32].
__device__ __forceinline__ void tdm_load_tile(const _Float16* gsrc, unsigned lds_off,
                                              int ncols_total, int K) {
    const unsigned long long ga = (unsigned long long)(size_t)gsrc;
    v4u g0;
    g0.x = 1u;                                                 // count=1 (valid user D#)
    g0.y = lds_off;                                            // lds_addr (bytes)
    g0.z = (unsigned)ga;                                       // global_addr[31:0]
    g0.w = (unsigned)((ga >> 32) & 0x1FFFFFFull) | (2u << 30); // addr[56:32] | type=2
    v8i g1;
    g1[0] = 1 << 16;                                   // data_size=1 (2 bytes)
    g1[1] = (K & 0xFFFF) << 16;                        // tensor_dim0 low16 (bits 63:48)
    g1[2] = ((K >> 16) & 0xFFFF) |                     // tensor_dim0 hi16
            ((ncols_total & 0xFFFF) << 16);            // tensor_dim1 low16
    g1[3] = ((ncols_total >> 16) & 0xFFFF) |           // tensor_dim1 hi16
            (32 << 16);                                // tile_dim0 = 32 (k extent)
    g1[4] = 64;                                        // tile_dim1 = 64 cols, tile_dim2=0
    g1[5] = K;                                         // tensor_dim0_stride low32 (elements)
    g1[6] = 0;                                         // stride hi16 | dim1_stride low16
    g1[7] = 0;
    v4i z4 = {0, 0, 0, 0};
#if __has_include(<hip/amd_detail/amd_gfx1250_TDM.h>)
    v8i z8 = {0, 0, 0, 0, 0, 0, 0, 0};
    __builtin_amdgcn_tensor_load_to_lds(g0, g1, z4, z4, z8, 0);
#else
    __builtin_amdgcn_tensor_load_to_lds(g0, g1, z4, z4, 0);
#endif
}
#endif

// ---------------------------------------------------------------- GEMM: Y = A(MxK) * W(NxK)^T + bias
// f16 in, fp32 WMMA accumulate, f16 out. One wave -> 16x64 tile; block = 8 waves -> 128x64.
// W panel (64 cols x 32 k) staged in LDS per block: TDM double-buffered if available.
__global__ void gemm_wmma_f16(const _Float16* __restrict__ A,
                              const _Float16* __restrict__ W,
                              const float* __restrict__ bias,
                              _Float16* __restrict__ out,
                              int M, int N, int K, int relu) {
    __shared__ __align__(16) _Float16 Bl[2][64 * 32];   // 2 x 4 KB

    const int lane = threadIdx.x & 31;
    const int wave = threadIdx.x >> 5;
    const int m0   = blockIdx.y * 128 + wave * 16;
    const int n0   = blockIdx.x * 64;

    const int arow = m0 + (lane & 15);
    const int klo  = (lane >> 4) * 16;    // A frag: k {0..15} / {16..31}
    const int bcol = lane & 15;
    const int kb   = (lane >> 4) * 16;    // B frag: k {0..15} / {16..31}
    const int ksteps = K >> 5;

    v8f acc0 = {}, acc1 = {}, acc2 = {}, acc3 = {};

#if HAVE_TDM
    // opaque guard store so LLVM can't treat TDM-written LDS as never-stored
    if (relu == 12345) { Bl[0][threadIdx.x] = (_Float16)0; Bl[1][threadIdx.x] = (_Float16)0; }
    if (wave == 0)
        tdm_load_tile(W + (size_t)n0 * K, (unsigned)(size_t)&Bl[0][0], N, K);
#endif

    for (int ks = 0; ks < ksteps; ++ks) {
        const int k = ks << 5;
#if HAVE_TDM
        const int bufi = ks & 1;
        if (wave == 0) {
            if (ks + 1 < ksteps) {
                tdm_load_tile(W + (size_t)n0 * K + (k + 32),
                              (unsigned)(size_t)&Bl[bufi ^ 1][0], N, K);
                __builtin_amdgcn_s_wait_tensorcnt(1);   // current buffer landed
            } else {
                __builtin_amdgcn_s_wait_tensorcnt(0);
            }
        }
        __syncthreads();                                 // LDS panel visible to all waves
#else
        const int bufi = 0;
        __syncthreads();                                 // prior reads done before overwrite
        {   // cooperative staging: 256 threads x 8 halves = 64x32 tile
            const int t   = threadIdx.x;
            const int col = t >> 2;
            const int kk  = (t & 3) * 8;
            *(v8h*)&Bl[0][col * 32 + kk] =
                *(const v8h*)(W + (size_t)(n0 + col) * K + k + kk);
        }
        __syncthreads();
#endif
        // A fragment: two 16-byte global loads per lane
        const _Float16* ap = A + (size_t)arow * K + k + klo;
        __builtin_prefetch(ap + 64, 0, 0);               // global_prefetch_b8, k-panel two steps ahead
        v8h alo = *(const v8h*)ap;
        v8h ahi = *(const v8h*)(ap + 8);
        v16h a;
#pragma unroll
        for (int j = 0; j < 8; ++j) { a[j] = alo[j]; a[8 + j] = ahi[j]; }

        // B fragments from LDS
        v16h b[4];
#pragma unroll
        for (int t = 0; t < 4; ++t) {
            const _Float16* bp = &Bl[bufi][(t * 16 + bcol) * 32 + kb];
            v8h b0 = *(const v8h*)bp;
            v8h b1 = *(const v8h*)(bp + 8);
#pragma unroll
            for (int j = 0; j < 8; ++j) { b[t][j] = b0[j]; b[t][8 + j] = b1[j]; }
        }
        acc0 = WMMA_F16(a, b[0], acc0);
        acc1 = WMMA_F16(a, b[1], acc1);
        acc2 = WMMA_F16(a, b[2], acc2);
        acc3 = WMMA_F16(a, b[3], acc3);
#if HAVE_TDM
        __syncthreads();                                 // reads done before next TDM issue
#endif
    }

    const int rbase = m0 + ((lane >> 4) << 3);           // C: M = r + 8*(lane>=16)
    v8f accs[4] = {acc0, acc1, acc2, acc3};
#pragma unroll
    for (int t = 0; t < 4; ++t) {
        const int col = n0 + t * 16 + bcol;
        const float bv = bias[col];
#pragma unroll
        for (int r = 0; r < 8; ++r) {
            float v = accs[t][r] + bv;
            if (relu) v = v > 0.f ? v : 0.f;
            out[(size_t)(rbase + r) * N + col] = (_Float16)v;
        }
    }
}

// ---------------------------------------------------------------- attention over batch axis (16)
// One wave per (s,h): scores 16x16 (K=96) -> softmax over keys -> ctx 16x96.
__global__ void attention_wmma(const _Float16* __restrict__ qkv,
                               _Float16* __restrict__ ctx) {
    __shared__ __align__(16) _Float16 Pl[8][256];   // per-wave 16x16 prob tile

    const int lane = threadIdx.x & 31;
    const int wave = threadIdx.x >> 5;
    const int idx  = blockIdx.x * 8 + wave;   // (s,h) pair
    const int s    = idx >> 3;
    const int h    = idx & 7;
    const int l15  = lane & 15;
    const int klo  = (lane >> 4) * 8;
    const int kb   = (lane >> 4) * 8;

    const size_t rs = 3 * EMB;   // qkv row stride = 2304
    const size_t qbase = ((size_t)l15 * SEQ + s) * rs + (size_t)h * DH;
    const size_t kbase = ((size_t)l15 * SEQ + s) * rs + EMB + (size_t)h * DH;

    // scores = Q * K^T  (16x16, K-dim = 96 = 3 x 32)
    v8f sc = {};
#pragma unroll
    for (int k = 0; k < DH; k += 32) {
        v8h alo = *(const v8h*)(qkv + qbase + k + klo);
        v8h ahi = *(const v8h*)(qkv + qbase + k + 16 + klo);
        v16h a;
#pragma unroll
        for (int j = 0; j < 8; ++j) { a[j] = alo[j]; a[8 + j] = ahi[j]; }
        v8h b0 = *(const v8h*)(qkv + kbase + k + kb);
        v8h b1 = *(const v8h*)(qkv + kbase + k + kb + 16);
        v16h b;
#pragma unroll
        for (int j = 0; j < 8; ++j) { b[j] = b0[j]; b[8 + j] = b1[j]; }
        sc = WMMA_F16(a, b, sc);
    }

    // softmax over keys (= across the 16 lanes of each row)
    const float scale = 0.10206207261596576f;  // 1/sqrt(96)
    float p[8];
#pragma unroll
    for (int r = 0; r < 8; ++r) {
        float x = sc[r] * scale;
        float m = x;
        m = fmaxf(m, SWZ_XOR_F(m, 1));
        m = fmaxf(m, SWZ_XOR_F(m, 2));
        m = fmaxf(m, SWZ_XOR_F(m, 4));
        m = fmaxf(m, SWZ_XOR_F(m, 8));
        float e = __expf(x - m);
        float t = e;
        t += SWZ_XOR_F(t, 1);
        t += SWZ_XOR_F(t, 2);
        t += SWZ_XOR_F(t, 4);
        t += SWZ_XOR_F(t, 8);
        p[r] = e / t;
    }

    // C-layout -> A-layout via LDS
    _Float16* Pw = Pl[wave];
#pragma unroll
    for (int r = 0; r < 8; ++r)
        Pw[(r + ((lane >> 4) << 3)) * 16 + l15] = (_Float16)p[r];
    __syncthreads();

    v8h plo = *(const v8h*)(Pw + l15 * 16 + klo);
    v16h pa;
#pragma unroll
    for (int j = 0; j < 8; ++j) { pa[j] = plo[j]; pa[8 + j] = (_Float16)0; }  // pad K 16..31

    // ctx = P(16x16) * V(16x96): 6 N-tiles of 16
    const size_t vbase = (size_t)s * rs + 2 * EMB + (size_t)h * DH;
#pragma unroll
    for (int t = 0; t < 6; ++t) {
        v16h vb;
        {   // elems 0..7: k = klo..klo+7 real; elems 8..15 (k 16..31) zero-padded
#pragma unroll
            for (int j = 0; j < 8; ++j) {
                vb[j] = qkv[vbase + (size_t)(klo + j) * SEQ * rs + t * 16 + l15];
                vb[8 + j] = (_Float16)0;
            }
        }
        v8f c = {};
        c = WMMA_F16(pa, vb, c);
#pragma unroll
        for (int r = 0; r < 8; ++r) {
            const int b = r + ((lane >> 4) << 3);
            ctx[((size_t)b * SEQ + s) * EMB + h * DH + t * 16 + l15] = (_Float16)c[r];
        }
    }
}

// ---------------------------------------------------------------- prob = h @ w2^T + b2 (N=1)
__global__ void prob_kernel(const _Float16* __restrict__ hbuf,
                            const float* __restrict__ w2,
                            const float* __restrict__ b2,
                            float* __restrict__ prob) {
    const int lane = threadIdx.x & 31;
    const int wave = threadIdx.x >> 5;
    const int row  = blockIdx.x * 8 + wave;
    float acc = 0.f;
    for (int j = lane; j < 384; j += 32)
        acc += (float)hbuf[(size_t)row * 384 + j] * w2[j];
    acc += SWZ_XOR_F(acc, 1);
    acc += SWZ_XOR_F(acc, 2);
    acc += SWZ_XOR_F(acc, 4);
    acc += SWZ_XOR_F(acc, 8);
    acc += SWZ_XOR_F(acc, 16);
    if (lane == 0) prob[row] = acc + b2[0];
}

// ---------------------------------------------------------------- contacts[b,i,:] = prob[b*S+i]
// Write-once 268 MB output: non-temporal stores keep it out of L2.
__global__ void broadcast_kernel(const float* __restrict__ prob,
                                 float* __restrict__ out) {
    const float p = prob[blockIdx.x];
    v4f v = {p, p, p, p};
    v4f* o = (v4f*)(out + (size_t)blockIdx.x * SEQ);
    for (int i = threadIdx.x; i < SEQ / 4; i += blockDim.x)
        __builtin_nontemporal_store(v, o + i);
}

// ---------------------------------------------------------------- launch
extern "C" void kernel_launch(void* const* d_in, const int* in_sizes, int n_in,
                              void* d_out, int out_size, void* d_ws, size_t ws_size,
                              hipStream_t stream) {
    const float* features   = (const float*)d_in[0];
    const float* in_proj_w  = (const float*)d_in[1];
    const float* in_proj_b  = (const float*)d_in[2];
    const float* out_proj_w = (const float*)d_in[3];
    const float* out_proj_b = (const float*)d_in[4];
    const float* w1         = (const float*)d_in[5];
    const float* b1         = (const float*)d_in[6];
    const float* w2         = (const float*)d_in[7];
    const float* b2         = (const float*)d_in[8];

    // workspace layout (bytes, 256-aligned); aliasing: X16->ctx, qkv->attn_out+h
    char* ws = (char*)d_ws;
    _Float16* Wq16  = (_Float16*)(ws + 0);                      //  3,538,944  (2304x768)
    _Float16* Wo16  = (_Float16*)(ws + 3538944);                //  1,179,648  (768x768)
    _Float16* W116  = (_Float16*)(ws + 4718592);                //    589,824  (384x768)
    float*    prob  = (float*)   (ws + 5308416);                //    131,072  (32768)
    _Float16* X16   = (_Float16*)(ws + 5439488);                // 50,331,648  features -> later ctx
    _Float16* QKV16 = (_Float16*)(ws + 55771136);               //150,994,944  qkv -> later attn_out
    _Float16* H16   = (_Float16*)(ws + 55771136 + 50331648);    // 25,165,824  mlp hidden

    // 1. cast inputs to f16
    cast_f32_to_f16<<<4096, 256, 0, stream>>>(features,   X16,  MROWS * EMB);
    cast_f32_to_f16<<<2048, 256, 0, stream>>>(in_proj_w,  Wq16, 3 * EMB * EMB);
    cast_f32_to_f16<<<1024, 256, 0, stream>>>(out_proj_w, Wo16, EMB * EMB);
    cast_f32_to_f16<<<512,  256, 0, stream>>>(w1,         W116, 384 * EMB);

    // 2. qkv = X @ Wq^T + b      (32768 x 2304)
    gemm_wmma_f16<<<dim3(2304 / 64, MROWS / 128), 256, 0, stream>>>(
        X16, Wq16, in_proj_b, QKV16, MROWS, 3 * EMB, EMB, 0);

    // 3. attention over batch axis -> ctx (reuses X16 region)
    attention_wmma<<<SEQ * HEADS / 8, 256, 0, stream>>>(QKV16, X16);

    // 4. attn_out = ctx @ Wo^T + b  (reuses qkv region)
    gemm_wmma_f16<<<dim3(768 / 64, MROWS / 128), 256, 0, stream>>>(
        X16, Wo16, out_proj_b, QKV16, MROWS, EMB, EMB, 0);

    // 5. h = relu(attn_out @ w1^T + b1)
    gemm_wmma_f16<<<dim3(384 / 64, MROWS / 128), 256, 0, stream>>>(
        QKV16, W116, b1, H16, MROWS, 384, EMB, 1);

    // 6. prob = h @ w2^T + b2
    prob_kernel<<<MROWS / 8, 256, 0, stream>>>(H16, w2, b2, prob);

    // 7. broadcast to (16, 2048, 2048)
    broadcast_kernel<<<MROWS, 256, 0, stream>>>(prob, (float*)d_out);
}
